// ParallelSlowCastle_10136122819071
// MI455X (gfx1250) — matrix operand order repaired and checked by reading, not compiled, not run on hardware
//
#include <hip/hip_runtime.h>
#include <hip/hip_bf16.h>

typedef __attribute__((ext_vector_type(16))) _Float16 v16h;
typedef __attribute__((ext_vector_type(8)))  float    v8f;
typedef __attribute__((ext_vector_type(4)))  int      v4i;

#define SCALE 0.125f              // 64^-0.5
#define NBH 16                    // b*h = 2*8
#define PROJ_ELEMS (2u*8u*1024u*64u)   // one projection buffer, f16 elems
#define HEAD_ELEMS (1024u*64u)

#if defined(__gfx1250__) && __has_builtin(__builtin_amdgcn_global_load_async_to_lds_b128) && __has_builtin(__builtin_amdgcn_s_wait_asynccnt)
#define HAVE_ASYNC 1
typedef __attribute__((address_space(1))) v4i as1_v4i;   // global int32x4
typedef __attribute__((address_space(3))) v4i as3_v4i;   // LDS int32x4
#else
#define HAVE_ASYNC 0
#endif

// ---- WMMA fragment loaders (contiguous k order shared by A and B, wave32) ----
// A: 16x32 f16 (MxK). lanes 0-15: M=lane, K = 0..15;
//                     lanes 16-31: M=lane-16, K = 16..31.
template <typename T>
__device__ inline v16h load_a(const T* __restrict__ src, int ld, int lane) {
  int m  = lane & 15;
  int kb = (lane >> 4) << 4;      // 0 or 16
  v16h a;
#pragma unroll
  for (int i = 0; i < 16; ++i) a[i] = (_Float16)src[m * ld + kb + i];
  return a;
}

// B: 32x16 f16 (KxN), source row-major [K x N] with leading dim ld.
// lanes 0-15: N=lane, halves = K 0..15; lanes 16-31: N=lane-16, halves = K 16..31.
template <typename T>
__device__ inline v16h load_b(const T* __restrict__ src, int ld, int lane) {
  int n  = lane & 15;
  int kb = (lane >> 4) << 4;      // 0 or 16
  v16h b;
#pragma unroll
  for (int i = 0; i < 16; ++i) b[i] = (_Float16)src[(kb + i) * ld + n];
  return b;
}

// B = Mt^T where Mt is row-major [N x K] (i.e. B[k][n] = src[n*ld + k]).
template <typename T>
__device__ inline v16h load_bt(const T* __restrict__ src, int ld, int lane) {
  int n  = lane & 15;
  int kb = (lane >> 4) << 4;      // 0 or 16
  v16h b;
#pragma unroll
  for (int i = 0; i < 16; ++i) b[i] = (_Float16)src[n * ld + kb + i];
  return b;
}

__device__ inline v8f wmma16(v16h a, v16h b, v8f c) {
  return __builtin_amdgcn_wmma_f32_16x16x32_f16(false, a, false, b, (short)0, c,
                                                false, false);
}

// Cooperatively stage W[k0..k0+31][n0..n0+63] (f32) into padded LDS tile.
// 8 KB per tile; 512 16-byte chunks across 128 threads. Async path uses the
// gfx1250 GLOBAL_LOAD_ASYNC_TO_LDS_B128 + ASYNCcnt machinery.
__device__ inline void stage_w_tile(const float* W, int k0, int n0,
                                    float (*bt)[68], int tid) {
#pragma unroll
  for (int c = 0; c < 4; ++c) {
    int idx = tid + c * 128;          // 0..511
    int r = idx >> 4, q = idx & 15;   // row 0..31, 16B chunk 0..15
#if HAVE_ASYNC
    __builtin_amdgcn_global_load_async_to_lds_b128(
        (as1_v4i*)(W + (size_t)(k0 + r) * 512 + n0 + q * 4),
        (as3_v4i*)&bt[r][q * 4], 0, 0);
#else
    *(float4*)&bt[r][q * 4] =
        *(const float4*)(W + (size_t)(k0 + r) * 512 + n0 + q * 4);
#endif
  }
#if HAVE_ASYNC
  __builtin_amdgcn_s_wait_asynccnt(0);
#endif
}

// ================= K1: six input projections x @ W -> f16 [b,h,n,d] ========
__global__ void proj_kernel(const float* __restrict__ x,
                            const float* w0, const float* w1, const float* w2,
                            const float* w3, const float* w4, const float* w5,
                            _Float16* __restrict__ out) {
  __shared__ float bt[32][68];                    // padded, 16B-aligned rows
  const int lane = threadIdx.x & 31;
  const int wave = threadIdx.x >> 5;
  const int n0 = blockIdx.x * 64;                 // output col tile
  const int m0 = blockIdx.y * 64 + wave * 16;     // row strip for this wave
  const int ws = blockIdx.z;
  const float* W = (ws == 0) ? w0 : (ws == 1) ? w1 : (ws == 2) ? w2
                 : (ws == 3) ? w3 : (ws == 4) ? w4 : w5;
  _Float16* dst = out + (size_t)ws * PROJ_ELEMS;

  v8f acc[4] = {};
  for (int k0 = 0; k0 < 512; k0 += 32) {
    __syncthreads();                              // protect prior-iter reads
    stage_w_tile(W, k0, n0, bt, threadIdx.x);
    __syncthreads();
    v16h a = load_a(x + (size_t)m0 * 512 + k0, 512, lane);
#pragma unroll
    for (int t = 0; t < 4; ++t) {
      v16h b = load_b(&bt[0][t * 16], 68, lane);
      acc[t] = wmma16(a, b, acc[t]);
    }
  }
  const int nl = lane & 15, mb = (lane >> 4) << 3;
#pragma unroll
  for (int t = 0; t < 4; ++t) {
#pragma unroll
    for (int r = 0; r < 8; ++r) {
      int m = m0 + mb + r;                // flat row in [0,2048)
      int c = n0 + t * 16 + nl;           // flat col in [0,512)
      int bb = m >> 10, ni = m & 1023, h = c >> 6, d = c & 63;
      dst[(((size_t)(bb * 8 + h) * 1024 + ni) << 6) + d] = (_Float16)acc[t][r];
    }
  }
}

// ====== K2: term1 = tril(q_c v_u^T)*s ; sig = sigmoid(strict-triu(q_u k_u^T)*s)
__global__ void tri_kernel(const _Float16* __restrict__ proj,
                           _Float16* __restrict__ term1,
                           _Float16* __restrict__ sig) {
  const int lane = threadIdx.x & 31;
  const int wave = threadIdx.x >> 5;
  const int bh = blockIdx.z;
  const int i0 = blockIdx.y * 64 + (wave >> 1) * 32;
  const int j0 = blockIdx.x * 64 + (wave & 1) * 32;
  const size_t head = (size_t)bh * HEAD_ELEMS;
  const _Float16* q_u = proj + 0 * (size_t)PROJ_ELEMS + head;
  const _Float16* k_u = proj + 1 * (size_t)PROJ_ELEMS + head;
  const _Float16* v_u = proj + 2 * (size_t)PROJ_ELEMS + head;
  const _Float16* q_c = proj + 3 * (size_t)PROJ_ELEMS + head;

  v8f t1[2][2] = {}, lu[2][2] = {};
#pragma unroll
  for (int k0 = 0; k0 < 64; k0 += 32) {
    v16h aqc[2], aqu[2], bv[2], bk[2];
#pragma unroll
    for (int m = 0; m < 2; ++m) {
      aqc[m] = load_a(q_c + (size_t)(i0 + 16 * m) * 64 + k0, 64, lane);
      aqu[m] = load_a(q_u + (size_t)(i0 + 16 * m) * 64 + k0, 64, lane);
    }
#pragma unroll
    for (int n = 0; n < 2; ++n) {
      bv[n] = load_bt(v_u + (size_t)(j0 + 16 * n) * 64 + k0, 64, lane);
      bk[n] = load_bt(k_u + (size_t)(j0 + 16 * n) * 64 + k0, 64, lane);
    }
#pragma unroll
    for (int m = 0; m < 2; ++m)
#pragma unroll
      for (int n = 0; n < 2; ++n) {
        t1[m][n] = wmma16(aqc[m], bv[n], t1[m][n]);
        lu[m][n] = wmma16(aqu[m], bk[n], lu[m][n]);
      }
  }
  const int nl = lane & 15, mb = (lane >> 4) << 3;
  const size_t base = (size_t)bh << 20;   // bh * 1024*1024
#pragma unroll
  for (int m = 0; m < 2; ++m)
#pragma unroll
    for (int n = 0; n < 2; ++n)
#pragma unroll
      for (int r = 0; r < 8; ++r) {
        int i = i0 + 16 * m + mb + r, j = j0 + 16 * n + nl;
        float tv = (j <= i) ? t1[m][n][r] * SCALE : 0.f;
        float lg = lu[m][n][r] * SCALE;
        float sg = (j > i) ? 1.f / (1.f + __expf(-lg)) : 0.f;
        term1[base + (size_t)i * 1024 + j] = (_Float16)tv;
        sig[base + (size_t)i * 1024 + j]   = (_Float16)sg;
      }
}

// ====== K3: scores = tril(q_c k_c^T)*s - silu(term1 @ sig^T), -inf above diag
// Wave tile: 32 (i) x 64 (k)  -> 8 WMMA per jt step from 6 fragment loads.
__global__ void scores_kernel(const _Float16* __restrict__ proj,
                              const _Float16* __restrict__ term1,
                              const _Float16* __restrict__ sig,
                              _Float16* __restrict__ scores) {
  const int lane = threadIdx.x & 31;
  const int wave = threadIdx.x >> 5;
  const int bh = blockIdx.z;
  const int i0 = blockIdx.y * 64 + (wave >> 1) * 32;
  const int c0 = blockIdx.x * 128 + (wave & 1) * 64;   // output column (k index)
  const size_t base = (size_t)bh << 20;
  const int nl = lane & 15, mb = (lane >> 4) << 3;

  if (c0 >= i0 + 32) {                 // tile entirely above diagonal -> -inf
#pragma unroll
    for (int m = 0; m < 2; ++m)
#pragma unroll
      for (int n = 0; n < 4; ++n)
#pragma unroll
        for (int r = 0; r < 8; ++r) {
          int i = i0 + 16 * m + mb + r, k = c0 + 16 * n + nl;
          scores[base + (size_t)i * 1024 + k] = (_Float16)(-__builtin_inff());
        }
    return;
  }

  // S_c = q_c @ k_c^T * scale over d=64
  const size_t head = (size_t)bh * HEAD_ELEMS;
  const _Float16* q_c = proj + 3 * (size_t)PROJ_ELEMS + head;
  const _Float16* k_c = proj + 4 * (size_t)PROJ_ELEMS + head;
  v8f sc[2][4] = {};
#pragma unroll
  for (int k0 = 0; k0 < 64; k0 += 32) {
    v16h a0 = load_a(q_c + (size_t)i0 * 64 + k0, 64, lane);
    v16h a1 = load_a(q_c + (size_t)(i0 + 16) * 64 + k0, 64, lane);
#pragma unroll
    for (int n = 0; n < 4; ++n) {
      v16h b = load_bt(k_c + (size_t)(c0 + 16 * n) * 64 + k0, 64, lane);
      sc[0][n] = wmma16(a0, b, sc[0][n]);
      sc[1][n] = wmma16(a1, b, sc[1][n]);
    }
  }

  // S_u[i,k] = sum_j term1[i,j]*sig[k,j]; nonzero only for j in (k, i]
  const _Float16* t1 = term1 + base;
  const _Float16* sg = sig + base;
  v8f su[2][4] = {};
  const int jt0 = c0 >> 5;
  const int jt1 = (i0 + 31) >> 5;
  for (int jt = jt0; jt <= jt1; ++jt) {
    const int j0 = jt * 32;
    v16h a0 = load_a(t1 + (size_t)i0 * 1024 + j0, 1024, lane);
    v16h a1 = load_a(t1 + (size_t)(i0 + 16) * 1024 + j0, 1024, lane);
#pragma unroll
    for (int n = 0; n < 4; ++n) {
      v16h b = load_bt(sg + (size_t)(c0 + 16 * n) * 1024 + j0, 1024, lane);
      su[0][n] = wmma16(a0, b, su[0][n]);
      su[1][n] = wmma16(a1, b, su[1][n]);
    }
  }

#pragma unroll
  for (int m = 0; m < 2; ++m)
#pragma unroll
    for (int n = 0; n < 4; ++n)
#pragma unroll
      for (int r = 0; r < 8; ++r) {
        int i = i0 + 16 * m + mb + r, k = c0 + 16 * n + nl;
        float fin;
        if (k <= i) {
          float s_u = su[m][n][r];
          fin = sc[m][n][r] * SCALE - s_u / (1.f + __expf(-s_u));   // - silu
        } else {
          fin = -__builtin_inff();
        }
        scores[base + (size_t)i * 1024 + k] = (_Float16)fin;
      }
}

// ================= K4: in-place row softmax over f16 scores ================
__global__ void softmax_kernel(_Float16* __restrict__ scores) {
  __shared__ float red[256];
  const int t = threadIdx.x;
  _Float16* s = scores + (size_t)blockIdx.x * 1024;
  float v[4], m = -__builtin_inff();
#pragma unroll
  for (int u = 0; u < 4; ++u) { v[u] = (float)s[t + u * 256]; m = fmaxf(m, v[u]); }
  red[t] = m; __syncthreads();
  for (int o = 128; o; o >>= 1) { if (t < o) red[t] = fmaxf(red[t], red[t + o]); __syncthreads(); }
  m = red[0]; __syncthreads();
  float e[4], sum = 0.f;
#pragma unroll
  for (int u = 0; u < 4; ++u) { e[u] = __expf(v[u] - m); sum += e[u]; }
  red[t] = sum; __syncthreads();
  for (int o = 128; o; o >>= 1) { if (t < o) red[t] += red[t + o]; __syncthreads(); }
  const float inv = 1.f / red[0];
#pragma unroll
  for (int u = 0; u < 4; ++u) s[t + u * 256] = (_Float16)(e[u] * inv);
}

// ============ K5: attn_out = P @ v_c, written as [b, n, h*64] f16 ==========
// Wave tile: 32 (i) x 64 (d) -> 8 WMMA per k step from 6 fragment loads.
__global__ void pv_kernel(const _Float16* __restrict__ proj,
                          const _Float16* __restrict__ scores,
                          _Float16* __restrict__ attn) {
  const int lane = threadIdx.x & 31;
  const int wave = threadIdx.x >> 5;
  const int bh = blockIdx.y;
  const int i0 = blockIdx.x * 128 + wave * 32;
  const _Float16* P   = scores + ((size_t)bh << 20);
  const _Float16* v_c = proj + 5 * (size_t)PROJ_ELEMS + (size_t)bh * HEAD_ELEMS;

  v8f acc[2][4] = {};
  const int ktmax = (i0 + 31) >> 5;        // P[i,k]==0 for k>i
  for (int kt = 0; kt <= ktmax; ++kt) {
    const int k0 = kt * 32;
    v16h a0 = load_a(P + (size_t)i0 * 1024 + k0, 1024, lane);
    v16h a1 = load_a(P + (size_t)(i0 + 16) * 1024 + k0, 1024, lane);
#pragma unroll
    for (int t = 0; t < 4; ++t) {
      v16h b = load_b(v_c + (size_t)k0 * 64 + t * 16, 64, lane);
      acc[0][t] = wmma16(a0, b, acc[0][t]);
      acc[1][t] = wmma16(a1, b, acc[1][t]);
    }
  }
  const int bb = bh >> 3, h = bh & 7;
  const int nl = lane & 15, mb = (lane >> 4) << 3;
#pragma unroll
  for (int m = 0; m < 2; ++m)
#pragma unroll
    for (int t = 0; t < 4; ++t)
#pragma unroll
      for (int r = 0; r < 8; ++r)
        attn[(size_t)(bb * 1024 + i0 + 16 * m + mb + r) * 512 + h * 64 + t * 16 + nl] =
            (_Float16)acc[m][t][r];
}

// ================= K6: out = attn @ w_out, f32 output ======================
__global__ void outproj_kernel(const _Float16* __restrict__ attn,
                               const float* __restrict__ w_out,
                               float* __restrict__ out) {
  __shared__ float bt[32][68];
  const int lane = threadIdx.x & 31;
  const int wave = threadIdx.x >> 5;
  const int n0 = blockIdx.x * 64;
  const int m0 = blockIdx.y * 64 + wave * 16;
  v8f acc[4] = {};
  for (int k0 = 0; k0 < 512; k0 += 32) {
    __syncthreads();
    stage_w_tile(w_out, k0, n0, bt, threadIdx.x);
    __syncthreads();
    v16h a = load_a(attn + (size_t)m0 * 512 + k0, 512, lane);
#pragma unroll
    for (int t = 0; t < 4; ++t) {
      v16h b = load_b(&bt[0][t * 16], 68, lane);
      acc[t] = wmma16(a, b, acc[t]);
    }
  }
  const int nl = lane & 15, mb = (lane >> 4) << 3;
#pragma unroll
  for (int t = 0; t < 4; ++t)
#pragma unroll
    for (int r = 0; r < 8; ++r)
      out[(size_t)(m0 + mb + r) * 512 + n0 + t * 16 + nl] = acc[t][r];
}

extern "C" void kernel_launch(void* const* d_in, const int* in_sizes, int n_in,
                              void* d_out, int out_size, void* d_ws, size_t ws_size,
                              hipStream_t stream) {
  const float* x     = (const float*)d_in[0];
  const float* w_q_u = (const float*)d_in[1];
  const float* w_k_u = (const float*)d_in[2];
  const float* w_v_u = (const float*)d_in[3];
  const float* w_q_c = (const float*)d_in[4];
  const float* w_k_c = (const float*)d_in[5];
  const float* w_v_c = (const float*)d_in[6];
  const float* w_out = (const float*)d_in[7];
  float* out = (float*)d_out;

  // Workspace layout (all f16), ~110 MB total -> fits MI455X 192 MB L2.
  char* ws = (char*)d_ws;
  _Float16* proj   = (_Float16*)(ws);                                   // 6 x 2 MB
  _Float16* term1  = (_Float16*)(ws + 6u * PROJ_ELEMS * 2u);            // 32 MB
  _Float16* sig    = (_Float16*)((char*)term1 + (size_t)NBH * 1024 * 1024 * 2);
  _Float16* scores = (_Float16*)((char*)sig   + (size_t)NBH * 1024 * 1024 * 2);
  _Float16* attn   = (_Float16*)((char*)scores+ (size_t)NBH * 1024 * 1024 * 2);

  proj_kernel<<<dim3(8, 32, 6), 128, 0, stream>>>(x, w_q_u, w_k_u, w_v_u,
                                                  w_q_c, w_k_c, w_v_c, proj);
  tri_kernel<<<dim3(16, 16, 16), 128, 0, stream>>>(proj, term1, sig);
  scores_kernel<<<dim3(8, 16, 16), 128, 0, stream>>>(proj, term1, sig, scores);
  softmax_kernel<<<dim3(16384), 256, 0, stream>>>(scores);
  pv_kernel<<<dim3(8, 16), 128, 0, stream>>>(proj, scores, attn);
  outproj_kernel<<<dim3(8, 32), 128, 0, stream>>>(attn, w_out, out);
}
